// bimpm_25469156065631
// MI455X (gfx1250) — compile-verified, matrix operand order checked by reading, not run on hardware
//
#include <hip/hip_runtime.h>
#include <hip/hip_bf16.h>
#include <math.h>

// ---------------------------------------------------------------------------
// BiMPM matching layer for MI455X (gfx1250, wave32, WMMA).
// Heavy einsums -> v_wmma_f32_16x16x32_bf16 (fp32 accumulate).
// H=300 is divisible by 4 => all 4-wide K/N chunks are fully valid or fully
// invalid, so loaders use uniform predicates + b128/b64 vector loads.
// ---------------------------------------------------------------------------

typedef __attribute__((ext_vector_type(16))) __bf16 v16bf;
typedef __attribute__((ext_vector_type(8)))  float  v8f;
typedef __attribute__((ext_vector_type(4)))  float  fl4;
typedef __attribute__((ext_vector_type(4)))  __bf16 bh4;

#define BB   64
#define SS   256
#define HH   300
#define LL   10
#define LDK  40          // padded K stride (elements) in LDS tiles; 80B rows -> 16B aligned
#define EPSF 1e-8f

// Monotone order-preserving float<->uint mapping for atomicMax on floats.
__device__ __forceinline__ unsigned encf(float x) {
  unsigned u = __float_as_uint(x);
  return (u & 0x80000000u) ? ~u : (u | 0x80000000u);
}
__device__ __forceinline__ float decf(unsigned k) {
  return (k & 0x80000000u) ? __uint_as_float(k & 0x7fffffffu) : __uint_as_float(~k);
}
// Reference _div: n / where(d > EPS, d, EPS)
__device__ __forceinline__ float divE(float n, float d) {
  return n / (d > EPSF ? d : EPSF);
}

union FragU { v16bf v; uint4 q[2]; };

__device__ __forceinline__ bh4 cvt4(fl4 x) {
  bh4 r;
  r[0] = (__bf16)x[0]; r[1] = (__bf16)x[1]; r[2] = (__bf16)x[2]; r[3] = (__bf16)x[3];
  return r;
}

// One K-step of the 128x128 block tile: 8 waves in a 2(M)x4(N) grid,
// each wave owns a 64x32 region = 4x2 subtiles of 16x16.
// LDS layout: sA[m][k], sB[n][k], stride LDK, K-chunk = 32.
// bf16 A-fragment layout (ISA 7.12.2): lane = M (&15), half = lane>>4,
//   elements 0..7  = K 8*half .. 8*half+7   (one 16B LDS read)
//   elements 8..15 = K 16+8*half .. +7      (one 16B LDS read)
// B mirrors it with lane = N.
__device__ __forceinline__ void wmma_step(const __bf16* sA, const __bf16* sB,
                                          v8f acc[4][2], int lane, int m0w, int n0w) {
  const int half = lane >> 4, lr = lane & 15;
  v16bf afr[4], bfr[2];
#pragma unroll
  for (int mi = 0; mi < 4; ++mi) {
    const __bf16* p = sA + (m0w + 16 * mi + lr) * LDK + 8 * half;
    FragU u; u.q[0] = *(const uint4*)p; u.q[1] = *(const uint4*)(p + 16);
    afr[mi] = u.v;
  }
#pragma unroll
  for (int ni = 0; ni < 2; ++ni) {
    const __bf16* p = sB + (n0w + 16 * ni + lr) * LDK + 8 * half;
    FragU u; u.q[0] = *(const uint4*)p; u.q[1] = *(const uint4*)(p + 16);
    bfr[ni] = u.v;
  }
#pragma unroll
  for (int mi = 0; mi < 4; ++mi)
#pragma unroll
    for (int ni = 0; ni < 2; ++ni)
      acc[mi][ni] = __builtin_amdgcn_wmma_f32_16x16x32_bf16(
          false, afr[mi], false, bfr[ni], (short)0, acc[mi][ni], false, false);
}

// ------------------------------- small kernels ------------------------------

__global__ void zero_u32(unsigned* __restrict__ p, long n) {
  long i = (long)blockIdx.x * blockDim.x + threadIdx.x;
  long st = (long)gridDim.x * blockDim.x;
  for (; i < n; i += st) p[i] = 0u;
}

// n4 = number of 4-element groups
__global__ void cvt_bf16(const float* __restrict__ s, __bf16* __restrict__ d, long n4) {
  long i = (long)blockIdx.x * blockDim.x + threadIdx.x;
  long st = (long)gridDim.x * blockDim.x;
  for (; i < n4; i += st) {
    fl4 x = *(const fl4*)(s + 4 * i);
    *(bh4*)(d + 4 * i) = cvt4(x);
  }
}

// final_mask[s,t] = exists b: s>=qlen[b] && t>=alen[b]  <=>  t >= g[s],
// g[s] = min over {b : qlen[b] <= s} of alen[b]  (else large)
__global__ void gmask_kernel(const int* __restrict__ ql, const int* __restrict__ al,
                             int* __restrict__ g) {
  int s = threadIdx.x;
  int best = 1 << 30;
  for (int b = 0; b < BB; ++b)
    if (ql[b] <= s && al[b] < best) best = al[b];
  g[s] = best;
}

// Per-row norms: wn[b,s,l] = ||v[b,s,:]*w[l,:]|| for l<10, wn[b,s,10] = ||v[b,s,:]||
__global__ __launch_bounds__(256) void norms_kernel(
    const float* __restrict__ qfw, const float* __restrict__ afw,
    const float* __restrict__ qbw, const float* __restrict__ abw,
    const float* __restrict__ w3, const float* __restrict__ w4,
    float* __restrict__ oqf, float* __restrict__ oaf,
    float* __restrict__ oqb, float* __restrict__ oab) {
  __shared__ float sW[LL * HH];
  const int which = blockIdx.y;
  const float* v = (which == 0) ? qfw : (which == 1) ? afw : (which == 2) ? qbw : abw;
  const float* w = (which < 2) ? w3 : w4;
  float* o = (which == 0) ? oqf : (which == 1) ? oaf : (which == 2) ? oqb : oab;
  for (int i = threadIdx.x; i < LL * HH; i += blockDim.x) sW[i] = w[i];
  __syncthreads();
  int idx = blockIdx.x * blockDim.x + threadIdx.x;  // b*SS+s
  float acc[11];
#pragma unroll
  for (int l = 0; l < 11; ++l) acc[l] = 0.f;
  const float* vp = v + (size_t)idx * HH;
  for (int h = 0; h < HH; h += 4) {
    fl4 x = *(const fl4*)(vp + h);
    fl4 x2 = x * x;
    acc[10] += x2[0] + x2[1] + x2[2] + x2[3];
#pragma unroll
    for (int l = 0; l < LL; ++l) {
#pragma unroll
      for (int j = 0; j < 4; ++j) {
        float ww = sW[l * HH + h + j];
        acc[l] += x2[j] * ww * ww;
      }
    }
  }
  float* op = o + (size_t)idx * 11;
#pragma unroll
  for (int l = 0; l < 11; ++l) op[l] = sqrtf(acc[l]);
}

// ------------------------- pairwise WMMA kernel -----------------------------
// blockIdx: x = tile(2x2 of 128x128), y = l (0..9 weighted, 10 = attention), z = b.
// C = (v1 .* w[l]^2) x v2^T ; epilogue: cosine + mask, then
//   l<10 : atomic-max row/col reductions (maxpool matching)
//   l==10: store bf16 attention matrix + atomic-add row/col sums.
__global__ __launch_bounds__(256) void pairwise_wmma_kernel(
    const float* __restrict__ v1, const __bf16* __restrict__ v2bf,
    const float* __restrict__ w, const float* __restrict__ wn1,
    const float* __restrict__ wn2, const int* __restrict__ g,
    unsigned* __restrict__ qmax_enc, unsigned* __restrict__ amax_enc,
    __bf16* __restrict__ attbf, float* __restrict__ denomQ, float* __restrict__ denomA) {
  __shared__ __attribute__((aligned(16))) __bf16 sA[128 * LDK];
  __shared__ __attribute__((aligned(16))) __bf16 sB[128 * LDK];
  const int tid = threadIdx.x;
  const int tm = blockIdx.x >> 1, tn = blockIdx.x & 1;
  const int l = blockIdx.y, b = blockIdx.z;
  const int m0 = tm * 128, n0 = tn * 128;
  const bool isatt = (l == LL);
  const int kc = (tid & 7) * 4, r0 = tid >> 3;
  const int wid = tid >> 5, lane = tid & 31;
  const int m0w = (wid >> 2) * 64, n0w = (wid & 3) * 32;

  v8f acc[4][2];
  v8f vz = {};
#pragma unroll
  for (int mi = 0; mi < 4; ++mi)
#pragma unroll
    for (int ni = 0; ni < 2; ++ni) acc[mi][ni] = vz;

  for (int ks = 0; ks < 10; ++ks) {   // K = 300 padded to 320
    const int k0 = ks * 32;
    const int kchunk = k0 + kc;       // 4-aligned; HH%4==0 -> chunk all-valid or all-invalid
    const bool kvalid = (kchunk < HH);
    fl4 wv = {1.f, 1.f, 1.f, 1.f};
    if (!isatt && kvalid) {
      fl4 t = *(const fl4*)(w + l * HH + kchunk);   // global_load_b128
      wv = t * t;
    }
#pragma unroll
    for (int rr = 0; rr < 4; ++rr) {
      const int m = r0 + 32 * rr;  // 0..127, serves as A row and B row
      bh4 za = {}, zb = {};
      if (kvalid) {
        fl4 x = *(const fl4*)(v1 + ((size_t)(b * SS + m0 + m)) * HH + kchunk);  // b128
        za = cvt4(x * wv);
        zb = *(const bh4*)(v2bf + ((size_t)(b * SS + n0 + m)) * HH + kchunk);   // b64
      }
      *(bh4*)&sA[m * LDK + kc] = za;
      *(bh4*)&sB[m * LDK + kc] = zb;
    }
    __syncthreads();
    wmma_step(sA, sB, acc, lane, m0w, n0w);
    __syncthreads();
  }

  // Epilogue. C/D layout: lane in 0..15 -> M = vgpr, lanes 16..31 -> M = vgpr+8; N = lane&15.
  const int half = lane >> 4, lr = lane & 15;
#pragma unroll
  for (int mi = 0; mi < 4; ++mi) {
#pragma unroll
    for (int ni = 0; ni < 2; ++ni) {
      const int tcol = n0 + n0w + 16 * ni + lr;  // answer index t
      float colred = isatt ? 0.0f : -3.0e38f;
#pragma unroll
      for (int r = 0; r < 8; ++r) {
        const int srow = m0 + m0w + 16 * mi + 8 * half + r;  // query index s
        float d = wn1[(b * SS + srow) * 11 + l] * wn2[(b * SS + tcol) * 11 + l];
        float c = divE(acc[mi][ni][r], d);
        if (tcol >= g[srow]) c = 0.0f;  // final_mask -> 0
        if (isatt) {
          attbf[((size_t)(b * SS + srow)) * SS + tcol] = (__bf16)c;
          float rs = c;
          rs += __shfl_xor(rs, 1, 32); rs += __shfl_xor(rs, 2, 32);
          rs += __shfl_xor(rs, 4, 32); rs += __shfl_xor(rs, 8, 32);
          if (lr == 0) atomicAdd(&denomQ[b * SS + srow], rs);
          colred += c;
        } else {
          float rm = c;
          rm = fmaxf(rm, __shfl_xor(rm, 1, 32)); rm = fmaxf(rm, __shfl_xor(rm, 2, 32));
          rm = fmaxf(rm, __shfl_xor(rm, 4, 32)); rm = fmaxf(rm, __shfl_xor(rm, 8, 32));
          if (lr == 0) atomicMax(&qmax_enc[(b * SS + srow) * LL + l], encf(rm));
          colred = fmaxf(colred, c);
        }
      }
      if (isatt) {
        colred += __shfl_xor(colred, 16, 32);
        if (lane < 16) atomicAdd(&denomA[b * SS + tcol], colred);
      } else {
        colred = fmaxf(colred, __shfl_xor(colred, 16, 32));
        if (lane < 16) atomicMax(&amax_enc[(b * SS + tcol) * LL + l], encf(colred));
      }
    }
  }
}

// ------------------------------ mean WMMA kernel ----------------------------
// mean[b,m,n] = (att(^T) @ V)[m,n] / clamp(denom[b,m]).  M=256,K=256,N=300(->384 tiled).
__global__ __launch_bounds__(256) void mean_wmma_kernel(
    const __bf16* __restrict__ attbf, const __bf16* __restrict__ vbf,
    const float* __restrict__ denom, __bf16* __restrict__ outbf, int atrans) {
  __shared__ __attribute__((aligned(16))) __bf16 sA[128 * LDK];
  __shared__ __attribute__((aligned(16))) __bf16 sB[128 * LDK];
  const int tid = threadIdx.x;
  const int tn = blockIdx.x % 3, tm = blockIdx.x / 3;
  const int b = blockIdx.y;
  const int m0 = tm * 128, n0 = tn * 128;
  const int kc = (tid & 7) * 4, r0 = tid >> 3;
  const int nc = (tid & 31) * 4, kk0 = tid >> 5;  // also used as mc/kk0 for atrans A
  const int wid = tid >> 5, lane = tid & 31;
  const int m0w = (wid >> 2) * 64, n0w = (wid & 3) * 32;

  v8f acc[4][2];
  v8f vz = {};
#pragma unroll
  for (int mi = 0; mi < 4; ++mi)
#pragma unroll
    for (int ni = 0; ni < 2; ++ni) acc[mi][ni] = vz;

  for (int ks = 0; ks < 8; ++ks) {  // K = 256, always in range
    const int k0 = ks * 32;
    if (atrans) {
      // A[m][k] = att[b, k, m0+m]: read 4 consecutive m (contiguous, b64),
      // transpose on LDS store.
#pragma unroll
      for (int kk = 0; kk < 4; ++kk) {
        const int k = kk0 + 8 * kk;  // 0..31
        bh4 vv = *(const bh4*)(attbf + ((size_t)(b * SS + k0 + k)) * SS + m0 + nc);
#pragma unroll
        for (int j = 0; j < 4; ++j) sA[(nc + j) * LDK + k] = vv[j];
      }
    } else {
      // A[m][k] = att[b, m0+m, k]: contiguous in k (b64).
#pragma unroll
      for (int rr = 0; rr < 4; ++rr) {
        const int m = r0 + 32 * rr;
        *(bh4*)&sA[m * LDK + kc] =
            *(const bh4*)(attbf + ((size_t)(b * SS + m0 + m)) * SS + k0 + kc);
      }
    }
    // B tile [n][k] = V[b, k, n0+n]: read 4 consecutive n (b64), transpose on store.
#pragma unroll
    for (int kk = 0; kk < 4; ++kk) {
      const int k = kk0 + 8 * kk;
      bh4 vv = {};
      if (n0 + nc < HH)  // HH%4==0 -> chunk all-valid or all-invalid
        vv = *(const bh4*)(vbf + ((size_t)(b * SS + k0 + k)) * HH + n0 + nc);
#pragma unroll
      for (int j = 0; j < 4; ++j) sB[(nc + j) * LDK + k] = vv[j];
    }
    __syncthreads();
    wmma_step(sA, sB, acc, lane, m0w, n0w);
    __syncthreads();
  }

  const int half = lane >> 4, lr = lane & 15;
#pragma unroll
  for (int mi = 0; mi < 4; ++mi)
#pragma unroll
    for (int ni = 0; ni < 2; ++ni) {
      const int n = n0 + n0w + 16 * ni + lr;
      if (n < HH) {
#pragma unroll
        for (int r = 0; r < 8; ++r) {
          const int m = m0 + m0w + 16 * mi + 8 * half + r;
          float d = denom[b * SS + m];
          outbf[((size_t)(b * SS + m)) * HH + n] = (__bf16)divE(acc[mi][ni][r], d);
        }
      }
    }
}

// --------------------- per-position multi-perspective cosine ----------------
// mode 0: v2 row = len[b]-1 (fp32 src); mode 1: row 0 (fp32 src); mode 2: row s (bf16 means)
__global__ __launch_bounds__(256) void mp_match_kernel(
    const float* __restrict__ v1, const float* __restrict__ v2f,
    const __bf16* __restrict__ v2b, const int* __restrict__ len, int mode,
    const float* __restrict__ w10, float* __restrict__ out, int rowbase, int colbase) {
  __shared__ float sW[LL * HH];
  for (int i = threadIdx.x; i < LL * HH; i += blockDim.x) sW[i] = w10[i];
  __syncthreads();
  const int idx = blockIdx.x * blockDim.x + threadIdx.x;  // b*SS+s
  const int b = idx / SS, s = idx % SS;
  int r2 = 0;
  if (mode == 0) { int t = len[b] - 1; r2 = t > 0 ? t : 0; }
  else if (mode == 2) r2 = s;
  const size_t base2 = ((size_t)(b * SS + r2)) * HH;
  float nn[LL], d1[LL], d2[LL];
#pragma unroll
  for (int l = 0; l < LL; ++l) { nn[l] = 0.f; d1[l] = 0.f; d2[l] = 0.f; }
  for (int h = 0; h < HH; h += 4) {
    fl4 x1 = *(const fl4*)(v1 + (size_t)idx * HH + h);
    fl4 x2;
    if (mode == 2) {
      bh4 t = *(const bh4*)(v2b + base2 + h);
      x2[0] = (float)t[0]; x2[1] = (float)t[1]; x2[2] = (float)t[2]; x2[3] = (float)t[3];
    } else {
      x2 = *(const fl4*)(v2f + base2 + h);
    }
#pragma unroll
    for (int l = 0; l < LL; ++l) {
#pragma unroll
      for (int j = 0; j < 4; ++j) {
        float ww = sW[l * HH + h + j];
        float a = ww * x1[j], c = ww * x2[j];
        nn[l] += a * c; d1[l] += a * a; d2[l] += c * c;
      }
    }
  }
  float* op = out + ((size_t)b * (2 * SS) + rowbase + s) * 60 + colbase;
#pragma unroll
  for (int l = 0; l < LL; ++l) {
    float d = sqrtf(d1[l]) * sqrtf(d2[l]);
    op[l] = divE(nn[l], d);
  }
}

// Decode the encoded atomic-max buffers into output columns 20..39.
__global__ void decode_kernel(const unsigned* __restrict__ qf, const unsigned* __restrict__ qb,
                              const unsigned* __restrict__ af, const unsigned* __restrict__ ab,
                              float* __restrict__ out) {
  const long per = (long)BB * SS * LL;
  long i = (long)blockIdx.x * blockDim.x + threadIdx.x;
  long st = (long)gridDim.x * blockDim.x;
  for (; i < 4 * per; i += st) {
    int arr = (int)(i / per);
    long rem = i % per;
    int b = (int)(rem / (SS * LL));
    int s = (int)((rem / LL) % SS);
    int l = (int)(rem % LL);
    const unsigned* src = arr == 0 ? qf : arr == 1 ? qb : arr == 2 ? af : ab;
    int row = (arr < 2) ? s : SS + s;
    int col = (arr & 1) ? 30 + l : 20 + l;
    out[((size_t)b * (2 * SS) + row) * 60 + col] = decf(src[rem]);
  }
}

// --------------------------------- launcher ---------------------------------

extern "C" void kernel_launch(void* const* d_in, const int* in_sizes, int n_in,
                              void* d_out, int out_size, void* d_ws, size_t ws_size,
                              hipStream_t stream) {
  const float* qfw = (const float*)d_in[0];
  const float* qbw = (const float*)d_in[1];
  const float* afw = (const float*)d_in[2];
  const float* abw = (const float*)d_in[3];
  const float* w1 = (const float*)d_in[4];
  const float* w2 = (const float*)d_in[5];
  const float* w3 = (const float*)d_in[6];
  const float* w4 = (const float*)d_in[7];
  const float* w5 = (const float*)d_in[8];
  const float* w6 = (const float*)d_in[9];
  const int* qlen = (const int*)d_in[10];
  const int* alen = (const int*)d_in[11];
  float* out = (float*)d_out;

  const size_t nBSH = (size_t)BB * SS * HH;
  const size_t nBSS = (size_t)BB * SS * SS;
  char* ws = (char*)d_ws;
  size_t off = 0;
  auto alloc = [&](size_t bytes) -> char* {
    char* p = ws + off;
    off = (off + bytes + 255) & ~(size_t)255;
    return p;
  };

  __bf16* qfw_bf = (__bf16*)alloc(nBSH * 2);
  __bf16* qbw_bf = (__bf16*)alloc(nBSH * 2);
  __bf16* afw_bf = (__bf16*)alloc(nBSH * 2);
  __bf16* abw_bf = (__bf16*)alloc(nBSH * 2);
  __bf16* attf = (__bf16*)alloc(nBSS * 2);
  __bf16* attb = (__bf16*)alloc(nBSS * 2);
  float* wn_qf = (float*)alloc((size_t)BB * SS * 11 * 4);
  float* wn_af = (float*)alloc((size_t)BB * SS * 11 * 4);
  float* wn_qb = (float*)alloc((size_t)BB * SS * 11 * 4);
  float* wn_ab = (float*)alloc((size_t)BB * SS * 11 * 4);
  float* dQf = (float*)alloc((size_t)BB * SS * 4);   // zeroed block starts here
  float* dAf = (float*)alloc((size_t)BB * SS * 4);
  float* dQb = (float*)alloc((size_t)BB * SS * 4);
  float* dAb = (float*)alloc((size_t)BB * SS * 4);
  unsigned* eQf = (unsigned*)alloc((size_t)BB * SS * LL * 4);
  unsigned* eQb = (unsigned*)alloc((size_t)BB * SS * LL * 4);
  unsigned* eAf = (unsigned*)alloc((size_t)BB * SS * LL * 4);
  unsigned* eAb = (unsigned*)alloc((size_t)BB * SS * LL * 4);
  char* zero_end = ws + off;
  __bf16* amean_f = (__bf16*)alloc(nBSH * 2);
  __bf16* qmean_f = (__bf16*)alloc(nBSH * 2);
  __bf16* amean_b = (__bf16*)alloc(nBSH * 2);
  __bf16* qmean_b = (__bf16*)alloc(nBSH * 2);
  int* g = (int*)alloc(SS * 4);

  // init accumulator buffers (denoms=0, encoded-max=0 == minus-infinity)
  long zwords = (long)((zero_end - (char*)dQf) / 4);
  zero_u32<<<2048, 256, 0, stream>>>((unsigned*)dQf, zwords);
  gmask_kernel<<<1, 256, 0, stream>>>(qlen, alen, g);

  cvt_bf16<<<2048, 256, 0, stream>>>(qfw, qfw_bf, (long)(nBSH / 4));
  cvt_bf16<<<2048, 256, 0, stream>>>(qbw, qbw_bf, (long)(nBSH / 4));
  cvt_bf16<<<2048, 256, 0, stream>>>(afw, afw_bf, (long)(nBSH / 4));
  cvt_bf16<<<2048, 256, 0, stream>>>(abw, abw_bf, (long)(nBSH / 4));

  norms_kernel<<<dim3(BB * SS / 256, 4), 256, 0, stream>>>(
      qfw, afw, qbw, abw, w3, w4, wn_qf, wn_af, wn_qb, wn_ab);

  // pairwise maxpool matching (l=0..9) + attention matrix (l=10)
  pairwise_wmma_kernel<<<dim3(4, LL + 1, BB), 256, 0, stream>>>(
      qfw, afw_bf, w3, wn_qf, wn_af, g, eQf, eAf, attf, dQf, dAf);
  pairwise_wmma_kernel<<<dim3(4, LL + 1, BB), 256, 0, stream>>>(
      qbw, abw_bf, w4, wn_qb, wn_ab, g, eQb, eAb, attb, dQb, dAb);

  // attentive means
  mean_wmma_kernel<<<dim3(6, BB), 256, 0, stream>>>(attf, afw_bf, dQf, amean_f, 0);
  mean_wmma_kernel<<<dim3(6, BB), 256, 0, stream>>>(attf, qfw_bf, dAf, qmean_f, 1);
  mean_wmma_kernel<<<dim3(6, BB), 256, 0, stream>>>(attb, abw_bf, dQb, amean_b, 0);
  mean_wmma_kernel<<<dim3(6, BB), 256, 0, stream>>>(attb, qbw_bf, dAb, qmean_b, 1);

  // full matching: cols 0-9 (fw, w1) and 10-19 (bw, w2)
  mp_match_kernel<<<64, 256, 0, stream>>>(qfw, afw, nullptr, alen, 0, w1, out, 0, 0);
  mp_match_kernel<<<64, 256, 0, stream>>>(qbw, abw, nullptr, nullptr, 1, w2, out, 0, 10);
  mp_match_kernel<<<64, 256, 0, stream>>>(afw, qfw, nullptr, qlen, 0, w1, out, SS, 0);
  mp_match_kernel<<<64, 256, 0, stream>>>(abw, qbw, nullptr, nullptr, 1, w2, out, SS, 10);
  // attentive matching: cols 40-49 (w5) and 50-59 (w6)
  mp_match_kernel<<<64, 256, 0, stream>>>(qfw, nullptr, amean_f, nullptr, 2, w5, out, 0, 40);
  mp_match_kernel<<<64, 256, 0, stream>>>(qbw, nullptr, amean_b, nullptr, 2, w6, out, 0, 50);
  mp_match_kernel<<<64, 256, 0, stream>>>(afw, nullptr, qmean_f, nullptr, 2, w5, out, SS, 40);
  mp_match_kernel<<<64, 256, 0, stream>>>(abw, nullptr, qmean_b, nullptr, 2, w6, out, SS, 50);

  // maxpool results: cols 20-29 (fw) and 30-39 (bw)
  decode_kernel<<<2560, 256, 0, stream>>>(eQf, eQb, eAf, eAb, out);
  (void)in_sizes; (void)n_in; (void)out_size; (void)ws_size;
}
